// GCNEncoder_88562225644058
// MI455X (gfx1250) — compile-verified, matrix operand order checked by reading, not run on hardware
//
#include <hip/hip_runtime.h>
#include <hip/hip_bf16.h>

typedef __attribute__((ext_vector_type(16))) _Float16 v16h;
typedef __attribute__((ext_vector_type(8)))  float    v8f;

// ---------------------------------------------------------------------------
// Degree / normalization kernels
// ---------------------------------------------------------------------------
__global__ void gcn_init_deg(float* __restrict__ deg, int n) {
    int i = blockIdx.x * blockDim.x + threadIdx.x;
    if (i < n) deg[i] = 1.0f;                      // self-loop contributes 1
}

__global__ void gcn_deg_accum(const long long* __restrict__ src,
                              float* __restrict__ deg, int E) {
    int i = blockIdx.x * blockDim.x + threadIdx.x;
    if (i < E) atomicAdd(deg + (int)src[i], 1.0f);
}

__global__ void gcn_dinv(const float* __restrict__ deg,
                         float* __restrict__ dinv, int n) {
    int i = blockIdx.x * blockDim.x + threadIdx.x;
    if (i < n) {
        float d = deg[i];
        dinv[i] = d > 0.0f ? rsqrtf(d) : 0.0f;
    }
}

// ---------------------------------------------------------------------------
// Pre-pack W[K x FOUT] (fp32, row-major) into WMMA B-fragment order (f16):
// fragment index (ktile, coltile) -> 32 lanes x 16 f16 contiguous (1 KB).
// Lane's 16 values follow the ISA 16-bit B register striping, so the GEMM
// loads each B fragment as one aligned 32-byte vector (2 x b128).
// ---------------------------------------------------------------------------
template<int K, int FOUT>
__global__ void gcn_pack_w(const float* __restrict__ W,
                           _Float16* __restrict__ Wp) {
    constexpr int NKT = K / 32, NCT = FOUT / 16;
    int t = blockIdx.x * blockDim.x + threadIdx.x;
    if (t >= NKT * NCT * 32) return;
    const int lane = t & 31;
    const int ct   = (t >> 5) % NCT;
    const int kt   = (t >> 5) / NCT;
    const int n    = lane & 15;
    const int hi   = lane >> 4;

    v16h v;
    #pragma unroll
    for (int r = 0; r < 8; ++r) {
        const int k = kt * 32 + ((r >> 2) << 4) + (hi << 3) + ((r & 3) << 1);
        v[2 * r]     = (_Float16)W[(size_t)k       * FOUT + ct * 16 + n];
        v[2 * r + 1] = (_Float16)W[(size_t)(k + 1) * FOUT + ct * 16 + n];
    }
    ((v16h*)Wp)[((size_t)kt * NCT + ct) * 32 + lane] = v;
}

// ---------------------------------------------------------------------------
// Fused WMMA GEMM:
//   a-input : optionally relu(A + bias_in) applied during fp32->f16 convert
//   OutH    = A' @ W                         (gather source for edge scatter)
//   OutSelf = dinv[row]^2 * OutH (+bias_out) (self-loop-initialized accum)
// One block = 16-row strip; one wave = one 16-column tile; B fragments come
// from the pre-packed f16 buffer (single 32B vector load per k-tile).
// ---------------------------------------------------------------------------
template<int K, int FOUT, bool RELU_BIAS_IN, bool BIAS_OUT>
__global__ __launch_bounds__(32 * (FOUT / 16))
void gcn_gemm_wmma(const float* __restrict__ A,
                   const _Float16* __restrict__ Wp,
                   const float* __restrict__ bias_in,
                   const float* __restrict__ dinv,
                   const float* __restrict__ bias_out,
                   float* __restrict__ OutH,
                   float* __restrict__ OutSelf,
                   int nRows) {
    constexpr int NCT = FOUT / 16;
    const int lane    = threadIdx.x & 31;
    const int wave    = threadIdx.x >> 5;          // column tile index
    const int rowBase = blockIdx.x * 16;
    const int mn      = lane & 15;                 // A: row id, B/C/D: col id
    const int hi      = lane >> 4;                 // lane half (0/1)
    const int colBase = wave * 16;

    const int aRow = min(rowBase + mn, nRows - 1); // clamp (guarded store)
    const float* __restrict__ arow = A + (size_t)aRow * K;
    const v16h* __restrict__ WpF   = (const v16h*)Wp;

    v8f c = {0.f, 0.f, 0.f, 0.f, 0.f, 0.f, 0.f, 0.f};

    #pragma unroll
    for (int kt = 0; kt < K; kt += 32) {
        v16h a;
        #pragma unroll
        for (int r = 0; r < 8; ++r) {
            // ISA 16-bit A layout: VGPR r<4 -> K half 0, r>=4 -> K half 1;
            // lane half adds 8; each VGPR packs K and K+1.
            const int k = kt + ((r >> 2) << 4) + (hi << 3) + ((r & 3) << 1);
            float2 av = *(const float2*)(arow + k);
            if constexpr (RELU_BIAS_IN) {
                av.x = fmaxf(av.x + bias_in[k],     0.0f);
                av.y = fmaxf(av.y + bias_in[k + 1], 0.0f);
            }
            a[2 * r]     = (_Float16)av.x;
            a[2 * r + 1] = (_Float16)av.y;
        }
        const v16h b = WpF[((size_t)(kt >> 5) * NCT + wave) * 32 + lane];
        c = __builtin_amdgcn_wmma_f32_16x16x32_f16(
                /*neg_a=*/false, a, /*neg_b=*/false, b,
                /*c_mod=*/(short)0, c, /*reuse_a=*/false, /*reuse_b=*/false);
    }

    #pragma unroll
    for (int v = 0; v < 8; ++v) {
        const int row = rowBase + v + hi * 8;      // C/D layout: VGPR v -> M
        if (row < nRows) {
            const size_t o = (size_t)row * FOUT + colBase + mn;
            const float h  = c[v];
            const float di = dinv[row];
            OutH[o] = h;
            if constexpr (BIAS_OUT)
                OutSelf[o] = di * di * h + bias_out[colBase + mn];
            else
                OutSelf[o] = di * di * h;
        }
    }
}

// ---------------------------------------------------------------------------
// Edge scatter: one wave per edge, VEC floats per lane (F = 32*VEC).
// Coalesced vector gather of Hin[src] (L2-resident table), scaled f32
// atomics into Out[dst] resolved at L2.
// ---------------------------------------------------------------------------
template<int VEC>
__global__ void gcn_edge_scatter(const long long* __restrict__ src,
                                 const long long* __restrict__ dst,
                                 const float* __restrict__ dinv,
                                 const float* __restrict__ Hin,
                                 float* __restrict__ Out, int E) {
    constexpr int F = 32 * VEC;
    const int gwave = (int)(((size_t)blockIdx.x * blockDim.x + threadIdx.x) >> 5);
    const int lane  = threadIdx.x & 31;
    if (gwave >= E) return;
    const int s = (int)src[gwave];
    const int d = (int)dst[gwave];
    const float norm = dinv[s] * dinv[d];
    const float* __restrict__ hs = Hin + (size_t)s * F + lane * VEC;
    float* __restrict__ od       = Out + (size_t)d * F + lane * VEC;

    float v[VEC];
    #pragma unroll
    for (int i = 0; i < VEC; ++i) v[i] = hs[i];
    #pragma unroll
    for (int i = 0; i < VEC; ++i) atomicAdd(od + i, norm * v[i]);
}

// ---------------------------------------------------------------------------
extern "C" void kernel_launch(void* const* d_in, const int* in_sizes, int n_in,
                              void* d_out, int out_size, void* d_ws, size_t ws_size,
                              hipStream_t stream) {
    const float*     x   = (const float*)d_in[0];
    const long long* ei  = (const long long*)d_in[1];
    const float*     W1  = (const float*)d_in[2];
    const float*     b1  = (const float*)d_in[3];
    const float*     W2  = (const float*)d_in[4];
    const float*     b2  = (const float*)d_in[5];
    float*           out = (float*)d_out;

    const int FIN = 256, FHID = 128, FOUT = 64;
    const int N = in_sizes[0] / FIN;        // 50000
    const int E = in_sizes[1] / 2;          // 1.6M
    const long long* src = ei;
    const long long* dst = ei + E;

    // Workspace layout:
    //   W1p[256*128 f16] | W2p[128*64 f16] | deg[N] | dinv[N] | T1[N*128] | T2[N*128]
    char* base = (char*)d_ws;
    _Float16* W1p = (_Float16*)base;                       // 64 KB
    _Float16* W2p = (_Float16*)(base + 64 * 1024);         // 16 KB
    float* deg    = (float*)(base + 80 * 1024);
    float* dinv   = deg + N;
    float* T1     = dinv + N;                              // N x 128 (reused as N x 64)
    float* T2     = T1 + (size_t)N * FHID;                 // N x 128

    const int TPB = 256;
    const int rowBlocks = (N + 15) / 16;

    // 1. degrees -> dinv
    gcn_init_deg <<<(N + TPB - 1) / TPB, TPB, 0, stream>>>(deg, N);
    gcn_deg_accum<<<(E + TPB - 1) / TPB, TPB, 0, stream>>>(src, deg, E);
    gcn_dinv     <<<(N + TPB - 1) / TPB, TPB, 0, stream>>>(deg, dinv, N);

    // 2. Pre-pack weights into WMMA B-fragment layout (f16)
    gcn_pack_w<256, 128><<<(8 * 8 * 32 + TPB - 1) / TPB, TPB, 0, stream>>>(W1, W1p);
    gcn_pack_w<128,  64><<<(4 * 4 * 32 + TPB - 1) / TPB, TPB, 0, stream>>>(W2, W2p);

    // 3. Layer-1 GEMM (fused self-loop init):  T1 = X@W1,  T2 = dinv^2 * T1
    gcn_gemm_wmma<256, 128, false, false><<<rowBlocks, 256, 0, stream>>>(
        x, W1p, nullptr, dinv, nullptr, T1, T2, N);

    // 4. Layer-1 edge scatter:  T2[dst] += norm * T1[src]
    gcn_edge_scatter<4><<<(((size_t)E * 32) + TPB - 1) / TPB, TPB, 0, stream>>>(
        src, dst, dinv, T1, T2, E);

    // 5. Layer-2 GEMM (fused relu(agg+b1) on input, self-loop + b2 on output):
    //    T1 = relu(T2+b1)@W2,  out = dinv^2 * T1 + b2
    gcn_gemm_wmma<128, 64, true, true><<<rowBlocks, 128, 0, stream>>>(
        T2, W2p, b1, dinv, b2, T1, out, N);

    // 6. Layer-2 edge scatter:  out[dst] += norm * T1[src]
    gcn_edge_scatter<2><<<(((size_t)E * 32) + TPB - 1) / TPB, TPB, 0, stream>>>(
        src, dst, dinv, T1, out, E);
}
